// RoformerAttention_76012331204816
// MI455X (gfx1250) — matrix operand order repaired and checked by reading, not compiled, not run on hardware
//
#include <hip/hip_runtime.h>

// ---------------------------------------------------------------------------
// Roformer attention for MI455X (gfx1250, wave32, WMMA).
// All GEMM stages use v_wmma_f32_16x16x32_bf16 (bf16 in, fp32 accumulate).
// ---------------------------------------------------------------------------

#define BB   8
#define SS   2048
#define EE   532
#define HH   2
#define HD   266
#define HALF 133
#define EP   544      // E padded to 34*16
#define HDP  288      // HD padded to 9*32
#define NTOK (BB*SS)  // 16384
#define SCALE_QK 0.061313906f   // 1/sqrt(266)

typedef __attribute__((ext_vector_type(16))) __bf16 v16bf;
typedef __attribute__((ext_vector_type(8)))  float  v8f;

union Frag {
  v16bf v;
  uint4 q[2];
  unsigned short u[16];
};

__device__ __forceinline__ unsigned short f2bf(float f) {
  unsigned int u = __float_as_uint(f);
  u += 0x7fffu + ((u >> 16) & 1u);          // round-to-nearest-even
  return (unsigned short)(u >> 16);
}

__device__ __forceinline__ v8f wmma_bf16(v16bf a, v16bf b, v8f c) {
  return __builtin_amdgcn_wmma_f32_16x16x32_bf16(false, a, false, b,
                                                 (short)0, c, false, false);
}

// Load B-fragment: B[k][n] with n = lane&15 selecting source row, 16
// contiguous bf16 (32B) along k starting at (lane>=16)*16.
__device__ __forceinline__ v16bf load_b_contig(const unsigned short* p) {
  Frag f;
  const uint4* q = (const uint4*)p;
  f.q[0] = q[0];
  f.q[1] = q[1];
  return f.v;
}

// ---------------------------------------------------------------------------
// K0: convert 4 weight matrices (532x532 f32) -> padded bf16 [4][544][544]
// ---------------------------------------------------------------------------
__global__ void k_convert_weights(const float* __restrict__ Wq,
                                  const float* __restrict__ Wk,
                                  const float* __restrict__ Wv,
                                  const float* __restrict__ Wo,
                                  unsigned short* __restrict__ Wb) {
  int idx = blockIdx.x * 256 + threadIdx.x;
  const int per = EE * EE;
  if (idx >= 4 * per) return;
  int mat = idx / per;
  int rem = idx - mat * per;
  int r = rem / EE, c = rem - r * EE;
  const float* W = (mat == 0) ? Wq : (mat == 1) ? Wk : (mat == 2) ? Wv : Wo;
  Wb[(size_t)mat * EP * EP + (size_t)r * EP + c] = f2bf(W[(size_t)r * EE + c]);
}

// ---------------------------------------------------------------------------
// K1: fused QKV projection + bias + RoPE + pack.
//   grid (NTOK/16, 3), block 128 (4 waves).  mat: 0=Q 1=K 2=V
//   x tile staged once into LDS as zero-padded bf16; A-fragments are two
//   unconditional ds_load_b128 per k-step.
//   Q,K -> bf16 [bh][s][HDP]   (rope'd)
//   V   -> bf16 [bh][HDP][S]   (transposed for attn@V B-operand)
// ---------------------------------------------------------------------------
__global__ void __launch_bounds__(128)
k_qkv_proj(const float* __restrict__ x,
           const unsigned short* __restrict__ Wb,
           const float* __restrict__ bq, const float* __restrict__ bk,
           const float* __restrict__ bv,
           unsigned short* __restrict__ Qb, unsigned short* __restrict__ Kb,
           unsigned short* __restrict__ Vt) {
  __shared__ unsigned short xs[16][EP];   // 17 KB bf16 x-tile (padded)
  __shared__ float sc[16][EP];            // 34 KB fp32 result strip

  const int rt   = blockIdx.x;
  const int mat  = blockIdx.y;
  const int wave = threadIdx.x >> 5;
  const int lane = threadIdx.x & 31;
  const int laneM  = lane & 15;
  const int laneHi = lane >> 4;
  const int kbase  = laneHi * 8;

  const unsigned short* Wmat = Wb + (size_t)mat * EP * EP;
  const float* bias = (mat == 0) ? bq : (mat == 1) ? bk : bv;

  // ---- stage x tile -> bf16 LDS (zero padded) ----
  for (int idx = threadIdx.x; idx < 16 * EP; idx += 128) {
    int m = idx / EP, j = idx - m * EP;
    float v = (j < EE) ? x[((size_t)rt * 16 + m) * EE + j] : 0.f;
    xs[m][j] = f2bf(v);
  }
  __syncthreads();

  for (int ct = wave; ct < EP / 16; ct += 4) {
    v8f acc = {0.f, 0.f, 0.f, 0.f, 0.f, 0.f, 0.f, 0.f};
    const int j = ct * 16 + laneM;           // output column (= W row)
    for (int kk = 0; kk < EP; kk += 32) {
      Frag a;
      a.q[0] = *(const uint4*)&xs[laneM][kk + 2 * kbase];
      a.q[1] = *(const uint4*)&xs[laneM][kk + 8 + 2 * kbase];
      v16bf b = load_b_contig(Wmat + (size_t)j * EP + kk + laneHi * 16);
      acc = wmma_bf16(a.v, b, acc);
    }
    float bj = (j < EE) ? bias[j] : 0.f;
#pragma unroll
    for (int r = 0; r < 8; ++r) {
      int m = r + 8 * laneHi;
      sc[m][j] = acc[r] + bj;
    }
  }
  __syncthreads();

  if (mat == 2) {
    // pack V transposed: Vt[bh][d][s]
    for (int idx = threadIdx.x; idx < 16 * EE; idx += 128) {
      int m = idx / EE, j = idx - m * EE;
      int rg = rt * 16 + m;
      int b = rg >> 11, s = rg & (SS - 1);
      int h = (j >= HD) ? 1 : 0;
      int d = j - h * HD;
      int bh = b * HH + h;
      Vt[((size_t)bh * HDP + d) * SS + s] = f2bf(sc[m][j]);
    }
  } else {
    unsigned short* dst = (mat == 0) ? Qb : Kb;
    // RoPE: pair (d, d+HALF) within each head
    for (int idx = threadIdx.x; idx < 16 * HD; idx += 128) {
      int m = idx / HD, p = idx - m * HD;
      int h = (p >= HALF) ? 1 : 0;
      int d = p - h * HALF;
      int rg = rt * 16 + m;
      int b = rg >> 11, s = rg & (SS - 1);
      int bh = b * HH + h;
      float t1 = sc[m][h * HD + d];
      float t2 = sc[m][h * HD + HALF + d];
      float freq = __expf(-9.210340372f * ((float)d * (1.0f / (float)HALF)));
      float ang  = (float)s * freq;
      float cv = __cosf(ang), sv = __sinf(ang);
      size_t base = ((size_t)bh * SS + s) * HDP;
      dst[base + d]        = f2bf(t1 * cv - t2 * sv);
      dst[base + HALF + d] = f2bf(t1 * sv + t2 * cv);
    }
  }
}

// ---------------------------------------------------------------------------
// K3: scores = Qr @ Kr^T * scale + mask*w   -> attn buffer (raw, pre-softmax)
//   grid (S/16, 16, B*H), block 32 (1 wave). Each block: 16x128 strip.
//   9 A fragments preloaded once, reused across 8 k-tiles.
// ---------------------------------------------------------------------------
__global__ void __launch_bounds__(32)
k_scores(const unsigned short* __restrict__ Qb,
         const unsigned short* __restrict__ Kb,
         const float* __restrict__ mask,
         const float* __restrict__ cdrw,
         float* __restrict__ attn) {
  const int qt = blockIdx.x;
  const int kc = blockIdx.y;
  const int bh = blockIdx.z;
  const int lane = threadIdx.x;
  const int laneM = lane & 15, laneHi = lane >> 4;
  const int kbase = laneHi * 8;
  const float cw = cdrw[0];

  Frag A[HDP / 32];
  {
    const unsigned short* qrow =
        Qb + ((size_t)bh * SS + qt * 16 + laneM) * HDP;
#pragma unroll
    for (int ks = 0; ks < HDP / 32; ++ks) {
      const unsigned short* p = qrow + ks * 32;
      A[ks].q[0] = *(const uint4*)(p + 2 * kbase);
      A[ks].q[1] = *(const uint4*)(p + 8 + 2 * kbase);
    }
  }

  for (int i = 0; i < 8; ++i) {
    const int kt = kc * 8 + i;
    v8f acc = {0.f, 0.f, 0.f, 0.f, 0.f, 0.f, 0.f, 0.f};
    const unsigned short* krow =
        Kb + ((size_t)bh * SS + kt * 16 + laneM) * HDP;
#pragma unroll
    for (int ks = 0; ks < HDP / 32; ++ks) {
      v16bf b = load_b_contig(krow + ks * 32 + laneHi * 16);
      acc = wmma_bf16(A[ks].v, b, acc);
    }
    const int ncol = kt * 16 + laneM;
#pragma unroll
    for (int r = 0; r < 8; ++r) {
      int m = r + 8 * laneHi;
      size_t off = ((size_t)bh * SS + qt * 16 + m) * SS + ncol;
      attn[off] = acc[r] * SCALE_QK + mask[off] * cw;
    }
  }
}

// ---------------------------------------------------------------------------
// K4: row softmax over attn (in place).  grid B*H*S, block 256.
// ---------------------------------------------------------------------------
__global__ void __launch_bounds__(256)
k_softmax(float* __restrict__ attn) {
  __shared__ float red[256];
  const int row = blockIdx.x;
  float* p = attn + (size_t)row * SS;
  const int tid = threadIdx.x;

  float v[8];
  float mx = -3.4e38f;
#pragma unroll
  for (int i = 0; i < 8; ++i) {
    v[i] = p[tid + i * 256];
    mx = fmaxf(mx, v[i]);
  }
  red[tid] = mx;
  __syncthreads();
  for (int s = 128; s > 0; s >>= 1) {
    if (tid < s) red[tid] = fmaxf(red[tid], red[tid + s]);
    __syncthreads();
  }
  mx = red[0];
  __syncthreads();

  float sum = 0.f;
#pragma unroll
  for (int i = 0; i < 8; ++i) {
    v[i] = __expf(v[i] - mx);
    sum += v[i];
  }
  red[tid] = sum;
  __syncthreads();
  for (int s = 128; s > 0; s >>= 1) {
    if (tid < s) red[tid] += red[tid + s];
    __syncthreads();
  }
  float inv = 1.0f / red[0];
#pragma unroll
  for (int i = 0; i < 8; ++i) p[tid + i * 256] = v[i] * inv;
}

// ---------------------------------------------------------------------------
// K5: ctx = attn @ V.  grid (6, S/16, B*H), block 32.
//   3 n-tiles (48 head-dim cols) per wave: the fp32->bf16 A conversion is
//   done once per k-step and amortized over 3 WMMAs (HDP/16 = 18 = 6*3).
// ---------------------------------------------------------------------------
__global__ void __launch_bounds__(32)
k_attn_v(const float* __restrict__ attn,
         const unsigned short* __restrict__ Vt,
         unsigned short* __restrict__ ctxb) {
  const int ng = blockIdx.x;          // 0..5, group of 3 n-tiles
  const int qt = blockIdx.y;
  const int bh = blockIdx.z;
  const int lane = threadIdx.x;
  const int laneM = lane & 15, laneHi = lane >> 4;

  const float* arow = attn + ((size_t)bh * SS + qt * 16 + laneM) * SS;
  const int d0 = ng * 48 + laneM;     // head-dim col of first tile (<288)
  const unsigned short* vrow = Vt + ((size_t)bh * HDP + d0) * SS;

  v8f acc0 = {0.f, 0.f, 0.f, 0.f, 0.f, 0.f, 0.f, 0.f};
  v8f acc1 = acc0, acc2 = acc0;
  for (int kk = 0; kk < SS; kk += 32) {
    Frag a;
    {
      const float4* p0 = (const float4*)(arow + kk + laneHi * 16);
      const float4* p1 = (const float4*)(arow + kk + 8 + laneHi * 16);
      float4 r0 = p0[0], r1 = p0[1], r2 = p1[0], r3 = p1[1];
      a.u[0] = f2bf(r0.x);  a.u[1] = f2bf(r0.y);
      a.u[2] = f2bf(r0.z);  a.u[3] = f2bf(r0.w);
      a.u[4] = f2bf(r1.x);  a.u[5] = f2bf(r1.y);
      a.u[6] = f2bf(r1.z);  a.u[7] = f2bf(r1.w);
      a.u[8]  = f2bf(r2.x); a.u[9]  = f2bf(r2.y);
      a.u[10] = f2bf(r2.z); a.u[11] = f2bf(r2.w);
      a.u[12] = f2bf(r3.x); a.u[13] = f2bf(r3.y);
      a.u[14] = f2bf(r3.z); a.u[15] = f2bf(r3.w);
    }
    const unsigned short* bp = vrow + kk + laneHi * 16;
    v16bf b0 = load_b_contig(bp);
    v16bf b1 = load_b_contig(bp + (size_t)16 * SS);
    v16bf b2 = load_b_contig(bp + (size_t)32 * SS);
    acc0 = wmma_bf16(a.v, b0, acc0);
    acc1 = wmma_bf16(a.v, b1, acc1);
    acc2 = wmma_bf16(a.v, b2, acc2);
  }

  const int h = bh & (HH - 1);
#pragma unroll
  for (int t = 0; t < 3; ++t) {
    int d = ng * 48 + t * 16 + laneM;
    v8f acc = (t == 0) ? acc0 : (t == 1) ? acc1 : acc2;
    if (d < HD) {
      int bcol = h * HD + d;
#pragma unroll
      for (int r = 0; r < 8; ++r) {
        int m = r + 8 * laneHi;
        int token = (bh >> 1) * SS + qt * 16 + m;
        ctxb[(size_t)token * EP + bcol] = f2bf(acc[r]);
      }
    }
  }
}

// ---------------------------------------------------------------------------
// K6: out = ctx @ Wo^T + bo.  grid (NTOK/16, EP/32), block 32.
//   2 column tiles per wave sharing the A fragments.
// ---------------------------------------------------------------------------
__global__ void __launch_bounds__(32)
k_out_proj(const unsigned short* __restrict__ ctxb,
           const unsigned short* __restrict__ Wob,
           const float* __restrict__ bo,
           float* __restrict__ out) {
  const int rt = blockIdx.x;
  const int ct = blockIdx.y;          // 0..16, pair of column tiles
  const int lane = threadIdx.x;
  const int laneM = lane & 15, laneHi = lane >> 4;
  const int kbase = laneHi * 8;

  const unsigned short* arow = ctxb + ((size_t)rt * 16 + laneM) * EP;
  const int j0 = ct * 32 + laneM;
  const int j1 = j0 + 16;
  const unsigned short* brow0 = Wob + (size_t)j0 * EP;
  const unsigned short* brow1 = Wob + (size_t)j1 * EP;

  v8f acc0 = {0.f, 0.f, 0.f, 0.f, 0.f, 0.f, 0.f, 0.f};
  v8f acc1 = acc0;
  for (int kk = 0; kk < EP; kk += 32) {
    Frag a;
    a.q[0] = *(const uint4*)(arow + kk + 2 * kbase);
    a.q[1] = *(const uint4*)(arow + kk + 8 + 2 * kbase);
    v16bf b0 = load_b_contig(brow0 + kk + laneHi * 16);
    v16bf b1 = load_b_contig(brow1 + kk + laneHi * 16);
    acc0 = wmma_bf16(a.v, b0, acc0);
    acc1 = wmma_bf16(a.v, b1, acc1);
  }

  if (j0 < EE) {
    float bj = bo[j0];
#pragma unroll
    for (int r = 0; r < 8; ++r) {
      int m = r + 8 * laneHi;
      out[((size_t)rt * 16 + m) * EE + j0] = acc0[r] + bj;
    }
  }
  if (j1 < EE) {
    float bj = bo[j1];
#pragma unroll
    for (int r = 0; r < 8; ++r) {
      int m = r + 8 * laneHi;
      out[((size_t)rt * 16 + m) * EE + j1] = acc1[r] + bj;
    }
  }
}

// ---------------------------------------------------------------------------
// launch
// ---------------------------------------------------------------------------
extern "C" void kernel_launch(void* const* d_in, const int* in_sizes, int n_in,
                              void* d_out, int out_size, void* d_ws,
                              size_t ws_size, hipStream_t stream) {
  const float* x    = (const float*)d_in[0];
  const float* mask = (const float*)d_in[1];
  const float* Wq   = (const float*)d_in[2];
  const float* bq   = (const float*)d_in[3];
  const float* Wk   = (const float*)d_in[4];
  const float* bk   = (const float*)d_in[5];
  const float* Wv   = (const float*)d_in[6];
  const float* bv   = (const float*)d_in[7];
  const float* Wo   = (const float*)d_in[8];
  const float* bo   = (const float*)d_in[9];
  const float* cdrw = (const float*)d_in[10];

  float* out  = (float*)d_out;
  float* attn = out + (size_t)BB * SS * EE;   // second output, flat

  // workspace layout (bytes)
  const size_t offW   = 0;
  const size_t szW    = (size_t)4 * EP * EP * 2;                 // 2.37 MB
  const size_t offQ   = offW + szW;
  const size_t szQK   = (size_t)BB * HH * SS * HDP * 2;          // 18.9 MB
  const size_t offK   = offQ + szQK;
  const size_t offV   = offK + szQK;
  const size_t offCtx = offV + szQK;
  const size_t szCtx  = (size_t)NTOK * EP * 2;                   // 17.8 MB
  const size_t needed = offCtx + szCtx;                          // ~76.8 MB

  char* ws = (char*)d_ws;
  unsigned short* Wb   = (unsigned short*)(ws + offW);
  unsigned short* Qb   = (unsigned short*)(ws + offQ);
  unsigned short* Kb   = (unsigned short*)(ws + offK);
  unsigned short* Vt   = (unsigned short*)(ws + offV);
  unsigned short* ctxb = (unsigned short*)(ws + offCtx);

  size_t zbytes = needed < ws_size ? needed : ws_size;
  hipMemsetAsync(d_ws, 0, zbytes, stream);   // zero pads (graph-capturable)

  // K0: weights -> bf16 padded
  {
    int total = 4 * EE * EE;
    k_convert_weights<<<(total + 255) / 256, 256, 0, stream>>>(Wq, Wk, Wv, Wo,
                                                               Wb);
  }
  // K1: QKV projection + RoPE + pack
  k_qkv_proj<<<dim3(NTOK / 16, 3), 128, 0, stream>>>(x, Wb, bq, bk, bv, Qb, Kb,
                                                     Vt);
  // K3: raw scores
  k_scores<<<dim3(SS / 16, 16, BB * HH), 32, 0, stream>>>(Qb, Kb, mask, cdrw,
                                                          attn);
  // K4: softmax in place
  k_softmax<<<BB * HH * SS, 256, 0, stream>>>(attn);
  // K5: attn @ V -> ctx (3 n-tiles per wave, 6*3*16 = 288 = HDP)
  k_attn_v<<<dim3(6, SS / 16, BB * HH), 32, 0, stream>>>(attn, Vt, ctxb);
  // K6: out projection (2 col tiles per wave, 17*32 = 544 = EP)
  k_out_proj<<<dim3(NTOK / 16, EP / 32), 32, 0, stream>>>(
      ctxb, Wb + (size_t)3 * EP * EP, bo, out);

  (void)in_sizes; (void)n_in; (void)out_size;
}
